// HIDMuseFormer_14061722927952
// MI455X (gfx1250) — matrix operand order repaired and checked by reading, not compiled, not run on hardware
//
#include <hip/hip_runtime.h>
#include <hip/hip_bf16.h>
#include <math.h>
#include <stdint.h>

typedef _Float16 h16v __attribute__((ext_vector_type(16)));
typedef _Float16 h8v  __attribute__((ext_vector_type(8)));
typedef _Float16 h4v  __attribute__((ext_vector_type(4)));
typedef float    f8v  __attribute__((ext_vector_type(8)));

// Problem constants (match reference setup_inputs)
#define B_  2
#define S_  1024
#define NB_ 64
#define V_  1024
#define D_  512
#define L_  12
#define H_  8
#define F_  2048
#define DH_ 64
#define T_  (NB_ + S_)   // 1088

// ---------------------------------------------------------------------------
// Generic batched WMMA GEMM:  C[M,N](f32) (+)= alpha * A[M,K] * Bt[N,K]^T
// Block = 128 threads (4 wave32), tile 64x64, K-step 32.
// B tiles are software-pipelined through double-buffered LDS using CDNA5
// async global->LDS DMA (ASYNCcnt, in-order completion -> s_wait_asynccnt 2
// releases the older tile while the next is still in flight).
// All M multiples of 64, N multiples of 64, K multiples of 32 (no guards).
// ---------------------------------------------------------------------------
template <typename TA>
__global__ __launch_bounds__(128)
void gemm_kernel(const TA* __restrict__ A, const _Float16* __restrict__ Bt,
                 float* __restrict__ C, int M, int N, int K, int ldc,
                 long long sAb, long long sAh, long long sBb, long long sBh,
                 long long sCb, long long sCh, int hdiv, float alpha, int accumulate)
{
    __shared__ _Float16 As[64][40];      // 80B row pitch -> b128-aligned fragments
    __shared__ _Float16 Bs[2][64][40];   // double-buffered async destination

    const int tid  = threadIdx.x;
    const int lane = tid & 31;
    const int w    = tid >> 5;
    const int mr   = lane & 15;
    const int hs   = lane >> 4;
    const int n0   = blockIdx.x * 64;
    const int m0   = blockIdx.y * 64;
    const int z    = blockIdx.z;

    const TA*       Ab = A  + (long long)(z / hdiv) * sAb + (long long)(z % hdiv) * sAh;
    const _Float16* Bb = Bt + (long long)(z / hdiv) * sBb + (long long)(z % hdiv) * sBh;
    float*          Cb = C  + (long long)(z / hdiv) * sCb + (long long)(z % hdiv) * sCh;

    f8v acc[4];
    for (int i = 0; i < 4; ++i)
        for (int j = 0; j < 8; ++j) acc[i][j] = 0.0f;

    // staging coordinates (uniform per thread across K-steps)
    const int ra4 = tid >> 3;          // f32 A path: row within 16-row group
    const int ca4 = (tid & 7) * 4;     // f32 A path: 4-float column
    const int ra8 = tid >> 2;          // f16 path: row within 32-row group
    const int ca8 = (tid & 3) * 8;     // f16 path: 8-half column

    // issue the async DMA for one 64x32 f16 B tile into LDS buffer `buf`
    auto issueB = [&](int k0, int buf) {
        for (int p = 0; p < 2; ++p) {
            int rr = p * 32 + ra8;
            const _Float16* gp = &Bb[(long long)(n0 + rr) * K + (k0 + ca8)];
            unsigned loff = (unsigned)(uintptr_t)&Bs[buf][rr][ca8];
            asm volatile("global_load_async_to_lds_b128 %0, %1, off"
                         :: "v"(loff), "v"(gp) : "memory");
        }
    };

    issueB(0, 0);                       // prologue: first B tile in flight

    const int ksteps = K >> 5;
    for (int kb = 0; kb < ksteps; ++kb) {
        const int k0 = kb << 5;
        const int buf = kb & 1;
        const bool has_next = (kb + 1) < ksteps;

        // ---- stage A tile: batch all global loads into registers first ----
        if constexpr (sizeof(TA) == 4) {
            const float* Af = (const float*)Ab;
            float4 av[4];
            for (int p = 0; p < 4; ++p)
                av[p] = *(const float4*)&Af[(long long)(m0 + p * 16 + ra4) * K + (k0 + ca4)];
            if (has_next) {
                issueB(k0 + 32, buf ^ 1);   // next B tile overlaps this tile's compute
                for (int p = 0; p < 4; ++p)
                    __builtin_prefetch(&Af[(long long)(m0 + p * 16 + ra4) * K + (k0 + 32 + ca4)], 0, 1);
            }
            for (int p = 0; p < 4; ++p) {
                h4v hv;
                hv[0] = (_Float16)av[p].x; hv[1] = (_Float16)av[p].y;
                hv[2] = (_Float16)av[p].z; hv[3] = (_Float16)av[p].w;
                *(h4v*)&As[p * 16 + ra4][ca4] = hv;
            }
        } else {
            const _Float16* Ah = (const _Float16*)Ab;
            h8v av[2];
            for (int p = 0; p < 2; ++p)
                av[p] = *(const h8v*)&Ah[(long long)(m0 + p * 32 + ra8) * K + (k0 + ca8)];
            if (has_next) {
                issueB(k0 + 32, buf ^ 1);
                for (int p = 0; p < 2; ++p)
                    __builtin_prefetch(&Ah[(long long)(m0 + p * 32 + ra8) * K + (k0 + 32 + ca8)], 0, 1);
            }
            for (int p = 0; p < 2; ++p)
                *(h8v*)&As[p * 32 + ra8][ca8] = av[p];
        }

        // release the tile we are about to read; the next tile may stay in flight
        if (has_next) asm volatile("s_wait_asynccnt 0x2" ::: "memory");
        else          asm volatile("s_wait_asynccnt 0x0" ::: "memory");
        __syncthreads();

        // A fragment: lane holds row (w*16+mr); K = hs*8+{0..7} then 16+hs*8+{0..7}
        h8v alo = *(const h8v*)&As[w * 16 + mr][hs * 8];
        h8v ahi = *(const h8v*)&As[w * 16 + mr][16 + hs * 8];
        h16v afrag = __builtin_shufflevector(alo, ahi,
            0,1,2,3,4,5,6,7,8,9,10,11,12,13,14,15);

        for (int nt = 0; nt < 4; ++nt) {
            // B fragment: lane holds column (nt*16+mr); K = hs*8+{0..7} then 16+hs*8+{0..7}
            h8v blo = *(const h8v*)&Bs[buf][nt * 16 + mr][hs * 8];
            h8v bhi = *(const h8v*)&Bs[buf][nt * 16 + mr][16 + hs * 8];
            h16v bfrag = __builtin_shufflevector(blo, bhi,
                0,1,2,3,4,5,6,7,8,9,10,11,12,13,14,15);
            acc[nt] = __builtin_amdgcn_wmma_f32_16x16x32_f16(
                false, afrag, false, bfrag, (short)0, acc[nt], false, false);
        }
        __syncthreads();
    }

    // C/D layout: VGPR v -> M = hs*8 + v ; lane column = mr
    for (int nt = 0; nt < 4; ++nt) {
        int n = n0 + nt * 16 + mr;
        for (int v = 0; v < 8; ++v) {
            int m = m0 + w * 16 + hs * 8 + v;
            long long idx = (long long)m * ldc + n;
            float val = alpha * acc[nt][v];
            if (accumulate) Cb[idx] += val; else Cb[idx] = val;
        }
    }
}

// ---------------------------------------------------------------------------
// Elementwise / helper kernels
// ---------------------------------------------------------------------------
__global__ void embed_kernel(const int* __restrict__ tok, const int* __restrict__ inst,
                             const float* __restrict__ tokE, const float* __restrict__ instE,
                             float* __restrict__ x, long long n)
{
    long long idx = (long long)blockIdx.x * blockDim.x + threadIdx.x;
    if (idx >= n) return;
    int d = (int)(idx % D_);
    long long bs = idx / D_;
    int t = tok[bs];
    int ii = inst[bs]; ii = ii < 0 ? 0 : (ii > 129 ? 129 : ii);
    x[idx] = tokE[(long long)t * D_ + d] + instE[(long long)ii * D_ + d];
}

__global__ void bar_kernel(const float* __restrict__ barE, float* __restrict__ sumx, long long n)
{
    long long idx = (long long)blockIdx.x * blockDim.x + threadIdx.x;
    if (idx >= n) return;
    int d  = (int)(idx % D_);
    int nb = (int)((idx / D_) % NB_);
    sumx[idx] = barE[(long long)nb * D_ + d];
}

// W [Lb,K,N] f32  ->  Wt [Lb,N,K] f16
__global__ void cvt_t_kernel(const float* __restrict__ W, _Float16* __restrict__ Wt,
                             int K, int N, long long total)
{
    long long idx = (long long)blockIdx.x * blockDim.x + threadIdx.x;
    if (idx >= total) return;
    int n = (int)(idx % N);
    long long r = idx / N;
    int k = (int)(r % K);
    long long l = r / K;
    Wt[l * (long long)K * N + (long long)n * K + k] = (_Float16)W[idx];
}

__global__ void cvt_kernel(const float* __restrict__ W, _Float16* __restrict__ Wt, long long n)
{
    long long idx = (long long)blockIdx.x * blockDim.x + threadIdx.x;
    if (idx < n) Wt[idx] = (_Float16)W[idx];
}

__global__ __launch_bounds__(128)
void ln_kernel(const float* __restrict__ in, float* __restrict__ out,
               const float* __restrict__ g, const float* __restrict__ b)
{
    __shared__ float red[128];
    const long long row = blockIdx.x;
    const float* x = in + row * D_;
    float s = 0.f;
    for (int i = threadIdx.x; i < D_; i += 128) s += x[i];
    red[threadIdx.x] = s; __syncthreads();
    for (int st = 64; st > 0; st >>= 1) {
        if ((int)threadIdx.x < st) red[threadIdx.x] += red[threadIdx.x + st];
        __syncthreads();
    }
    float mean = red[0] / D_; __syncthreads();
    float v = 0.f;
    for (int i = threadIdx.x; i < D_; i += 128) { float d = x[i] - mean; v += d * d; }
    red[threadIdx.x] = v; __syncthreads();
    for (int st = 64; st > 0; st >>= 1) {
        if ((int)threadIdx.x < st) red[threadIdx.x] += red[threadIdx.x + st];
        __syncthreads();
    }
    float rstd = rsqrtf(red[0] / D_ + 1e-5f);
    for (int i = threadIdx.x; i < D_; i += 128)
        out[row * D_ + i] = (x[i] - mean) * rstd * g[i] + b[i];
}

// src f32 [B,M,D] -> rope'd f16 dst [B,H,Tdst,DH] at time-offset tOff, pos = m
__global__ void rope_scatter_kernel(const float* __restrict__ src, _Float16* __restrict__ dst,
                                    int M, int tOff, int Tdst, long long total)
{
    long long idx = (long long)blockIdx.x * blockDim.x + threadIdx.x;
    if (idx >= total) return;
    int da = (int)(idx % D_);
    long long r = idx / D_;
    int m = (int)(r % M);
    int b = (int)(r / M);
    int h = da / DH_, d = da % DH_, j = d & 31;
    const float* p = src + ((long long)b * M + m) * D_ + h * DH_;
    float x1 = p[j], x2 = p[32 + j];
    float fr = (float)m * powf(10000.0f, -(float)j / 32.0f);
    float c = cosf(fr), s = sinf(fr);
    float val = (d < 32) ? (x1 * c - x2 * s) : (x1 * s + x2 * c);
    dst[(((long long)b * H_ + h) * Tdst + (tOff + m)) * DH_ + d] = (_Float16)val;
}

// src f32 [B,M,D] -> f16 dst V^T [B,H,DH,T_] at time-offset tOff
__global__ void vT_scatter_kernel(const float* __restrict__ src, _Float16* __restrict__ dst,
                                  int M, int tOff, long long total)
{
    long long idx = (long long)blockIdx.x * blockDim.x + threadIdx.x;
    if (idx >= total) return;
    int da = (int)(idx % D_);
    long long r = idx / D_;
    int m = (int)(r % M);
    int b = (int)(r / M);
    int h = da / DH_, d = da % DH_;
    dst[(((long long)b * H_ + h) * DH_ + d) * T_ + (tOff + m)] =
        (_Float16)src[((long long)b * M + m) * D_ + da];
}

__device__ __forceinline__ bool reg_mask(const int* chord, int b, int q, int cq, int j) {
    if (j < NB_) return j < cq;
    int s2 = j - NB_;
    return (chord[b * S_ + s2] == cq) && (q >= s2);
}
__device__ __forceinline__ bool sum_mask(const int* chord, int b, int q, int j) {
    if (j < NB_) return j <= q;
    return chord[b * S_ + (j - NB_)] == q;
}

// scores f32 [B,H,M,T] -> masked softmax -> probs f16 (same layout)
__global__ __launch_bounds__(128)
void softmax_reg_kernel(const float* __restrict__ sc, _Float16* __restrict__ p,
                        const int* __restrict__ chord)
{
    __shared__ float red[128];
    int q = blockIdx.x, h = blockIdx.y, b = blockIdx.z;
    long long base = (((long long)(b * H_ + h)) * S_ + q) * T_;
    int cq = chord[b * S_ + q];
    float vals[9];
    int cnt = 0;
    float mx = -3.4e38f;
    for (int j = threadIdx.x; j < T_; j += 128) {
        float v = reg_mask(chord, b, q, cq, j) ? sc[base + j] : -1e9f;
        vals[cnt++] = v;
        mx = fmaxf(mx, v);
    }
    red[threadIdx.x] = mx; __syncthreads();
    for (int st = 64; st > 0; st >>= 1) {
        if ((int)threadIdx.x < st) red[threadIdx.x] = fmaxf(red[threadIdx.x], red[threadIdx.x + st]);
        __syncthreads();
    }
    mx = red[0]; __syncthreads();
    float sum = 0.f;
    for (int i = 0; i < cnt; ++i) { vals[i] = expf(vals[i] - mx); sum += vals[i]; }
    red[threadIdx.x] = sum; __syncthreads();
    for (int st = 64; st > 0; st >>= 1) {
        if ((int)threadIdx.x < st) red[threadIdx.x] += red[threadIdx.x + st];
        __syncthreads();
    }
    float inv = 1.0f / red[0];
    cnt = 0;
    for (int j = threadIdx.x; j < T_; j += 128)
        p[base + j] = (_Float16)(vals[cnt++] * inv);
}

__global__ __launch_bounds__(128)
void softmax_sum_kernel(const float* __restrict__ sc, _Float16* __restrict__ p,
                        const int* __restrict__ chord)
{
    __shared__ float red[128];
    int q = blockIdx.x, h = blockIdx.y, b = blockIdx.z;
    long long base = (((long long)(b * H_ + h)) * NB_ + q) * T_;
    float vals[9];
    int cnt = 0;
    float mx = -3.4e38f;
    for (int j = threadIdx.x; j < T_; j += 128) {
        float v = sum_mask(chord, b, q, j) ? sc[base + j] : -1e9f;
        vals[cnt++] = v;
        mx = fmaxf(mx, v);
    }
    red[threadIdx.x] = mx; __syncthreads();
    for (int st = 64; st > 0; st >>= 1) {
        if ((int)threadIdx.x < st) red[threadIdx.x] = fmaxf(red[threadIdx.x], red[threadIdx.x + st]);
        __syncthreads();
    }
    mx = red[0]; __syncthreads();
    float sum = 0.f;
    for (int i = 0; i < cnt; ++i) { vals[i] = expf(vals[i] - mx); sum += vals[i]; }
    red[threadIdx.x] = sum; __syncthreads();
    for (int st = 64; st > 0; st >>= 1) {
        if ((int)threadIdx.x < st) red[threadIdx.x] += red[threadIdx.x + st];
        __syncthreads();
    }
    float inv = 1.0f / red[0];
    cnt = 0;
    for (int j = threadIdx.x; j < T_; j += 128)
        p[base + j] = (_Float16)(vals[cnt++] * inv);
}

__global__ void silu_mul_kernel(float* __restrict__ h1, const float* __restrict__ h3, long long n)
{
    long long idx = (long long)blockIdx.x * blockDim.x + threadIdx.x;
    if (idx >= n) return;
    float a = h1[idx];
    h1[idx] = (a / (1.0f + expf(-a))) * h3[idx];
}

// ---------------------------------------------------------------------------
// Host orchestration
// ---------------------------------------------------------------------------
extern "C" void kernel_launch(void* const* d_in, const int* in_sizes, int n_in,
                              void* d_out, int out_size, void* d_ws, size_t ws_size,
                              hipStream_t stream)
{
    const int*   tok   = (const int*)d_in[0];
    const int*   chord = (const int*)d_in[1];
    const int*   inst  = (const int*)d_in[2];
    const float* tokE  = (const float*)d_in[4];
    const float* instE = (const float*)d_in[5];
    const float* barE  = (const float*)d_in[6];
    const float* ln_as_g = (const float*)d_in[7];
    const float* ln_as_b = (const float*)d_in[8];
    const float* ln_ar_g = (const float*)d_in[9];
    const float* ln_ar_b = (const float*)d_in[10];
    const float* ln_fs_g = (const float*)d_in[11];
    const float* ln_fs_b = (const float*)d_in[12];
    const float* ln_fr_g = (const float*)d_in[13];
    const float* ln_fr_b = (const float*)d_in[14];
    const float* Wq = (const float*)d_in[15];
    const float* Wk = (const float*)d_in[16];
    const float* Wv = (const float*)d_in[17];
    const float* Wo = (const float*)d_in[18];
    const float* fs_w1 = (const float*)d_in[19];
    const float* fs_w3 = (const float*)d_in[20];
    const float* fs_w2 = (const float*)d_in[21];
    const float* fr_w1 = (const float*)d_in[22];
    const float* fr_w3 = (const float*)d_in[23];
    const float* fr_w2 = (const float*)d_in[24];
    const float* out_g = (const float*)d_in[25];
    const float* out_b = (const float*)d_in[26];
    float* out = (float*)d_out;

    size_t off = 0;
    auto carve = [&](size_t bytes) -> void* {
        void* p = (char*)d_ws + off;
        off += (bytes + 255) & ~(size_t)255;
        return p;
    };

    const long long DD  = (long long)D_ * D_;
    const long long DF  = (long long)D_ * F_;

    _Float16* WqT   = (_Float16*)carve((size_t)L_ * DD * 2);
    _Float16* WkT   = (_Float16*)carve((size_t)L_ * DD * 2);
    _Float16* WvT   = (_Float16*)carve((size_t)L_ * DD * 2);
    _Float16* WoT   = (_Float16*)carve((size_t)L_ * DD * 2);
    _Float16* fsw1T = (_Float16*)carve((size_t)L_ * DF * 2);
    _Float16* fsw3T = (_Float16*)carve((size_t)L_ * DF * 2);
    _Float16* fsw2T = (_Float16*)carve((size_t)L_ * DF * 2);
    _Float16* frw1T = (_Float16*)carve((size_t)L_ * DF * 2);
    _Float16* frw3T = (_Float16*)carve((size_t)L_ * DF * 2);
    _Float16* frw2T = (_Float16*)carve((size_t)L_ * DF * 2);
    _Float16* tokH  = (_Float16*)carve((size_t)V_ * D_ * 2);

    float* x    = (float*)carve((size_t)B_ * S_ * D_ * 4);
    float* sumx = (float*)carve((size_t)B_ * NB_ * D_ * 4);
    float* lnR  = (float*)carve((size_t)B_ * S_ * D_ * 4);
    float* lnS  = (float*)carve((size_t)B_ * NB_ * D_ * 4);
    float* pRq  = (float*)carve((size_t)B_ * S_ * D_ * 4);
    float* pRk  = (float*)carve((size_t)B_ * S_ * D_ * 4);
    float* pRv  = (float*)carve((size_t)B_ * S_ * D_ * 4);
    float* pSq  = (float*)carve((size_t)B_ * NB_ * D_ * 4);
    float* pSk  = (float*)carve((size_t)B_ * NB_ * D_ * 4);
    float* pSv  = (float*)carve((size_t)B_ * NB_ * D_ * 4);
    _Float16* qrH = (_Float16*)carve((size_t)B_ * H_ * S_ * DH_ * 2);
    _Float16* qsH = (_Float16*)carve((size_t)B_ * H_ * NB_ * DH_ * 2);
    _Float16* kH  = (_Float16*)carve((size_t)B_ * H_ * T_ * DH_ * 2);
    _Float16* vTH = (_Float16*)carve((size_t)B_ * H_ * DH_ * T_ * 2);
    float*    scR = (float*)carve((size_t)B_ * H_ * S_ * T_ * 4);
    _Float16* pbR = (_Float16*)carve((size_t)B_ * H_ * S_ * T_ * 2);
    float*    scS = (float*)carve((size_t)B_ * H_ * NB_ * T_ * 4);
    _Float16* pbS = (_Float16*)carve((size_t)B_ * H_ * NB_ * T_ * 2);
    float* aR  = (float*)carve((size_t)B_ * S_ * D_ * 4);
    float* aS  = (float*)carve((size_t)B_ * NB_ * D_ * 4);
    float* f1R = (float*)carve((size_t)B_ * S_ * F_ * 4);
    float* f3R = (float*)carve((size_t)B_ * S_ * F_ * 4);
    float* f1S = (float*)carve((size_t)B_ * NB_ * F_ * 4);
    float* f3S = (float*)carve((size_t)B_ * NB_ * F_ * 4);

    auto blocks = [](long long n) { return (unsigned)((n + 255) / 256); };

    auto gemmF = [&](const float* A, const _Float16* Bt, float* C,
                     int M, int N, int K, int ldc,
                     long long sAb, long long sAh, long long sBb, long long sBh,
                     long long sCb, long long sCh, int hdiv,
                     float alpha, int accum, int batch) {
        dim3 g(N / 64, M / 64, batch);
        gemm_kernel<float><<<g, 128, 0, stream>>>(A, Bt, C, M, N, K, ldc,
            sAb, sAh, sBb, sBh, sCb, sCh, hdiv, alpha, accum);
    };
    auto gemmH = [&](const _Float16* A, const _Float16* Bt, float* C,
                     int M, int N, int K, int ldc,
                     long long sAb, long long sAh, long long sBb, long long sBh,
                     long long sCb, long long sCh, int hdiv,
                     float alpha, int accum, int batch) {
        dim3 g(N / 64, M / 64, batch);
        gemm_kernel<_Float16><<<g, 128, 0, stream>>>(A, Bt, C, M, N, K, ldc,
            sAb, sAh, sBb, sBh, sCb, sCh, hdiv, alpha, accum);
    };

    // ---- weight conversion (f32 -> f16 transposed) ----
    cvt_t_kernel<<<blocks((long long)L_ * DD), 256, 0, stream>>>(Wq, WqT, D_, D_, (long long)L_ * DD);
    cvt_t_kernel<<<blocks((long long)L_ * DD), 256, 0, stream>>>(Wk, WkT, D_, D_, (long long)L_ * DD);
    cvt_t_kernel<<<blocks((long long)L_ * DD), 256, 0, stream>>>(Wv, WvT, D_, D_, (long long)L_ * DD);
    cvt_t_kernel<<<blocks((long long)L_ * DD), 256, 0, stream>>>(Wo, WoT, D_, D_, (long long)L_ * DD);
    cvt_t_kernel<<<blocks((long long)L_ * DF), 256, 0, stream>>>(fs_w1, fsw1T, D_, F_, (long long)L_ * DF);
    cvt_t_kernel<<<blocks((long long)L_ * DF), 256, 0, stream>>>(fs_w3, fsw3T, D_, F_, (long long)L_ * DF);
    cvt_t_kernel<<<blocks((long long)L_ * DF), 256, 0, stream>>>(fs_w2, fsw2T, F_, D_, (long long)L_ * DF);
    cvt_t_kernel<<<blocks((long long)L_ * DF), 256, 0, stream>>>(fr_w1, frw1T, D_, F_, (long long)L_ * DF);
    cvt_t_kernel<<<blocks((long long)L_ * DF), 256, 0, stream>>>(fr_w3, frw3T, D_, F_, (long long)L_ * DF);
    cvt_t_kernel<<<blocks((long long)L_ * DF), 256, 0, stream>>>(fr_w2, frw2T, F_, D_, (long long)L_ * DF);
    cvt_kernel<<<blocks((long long)V_ * D_), 256, 0, stream>>>(tokE, tokH, (long long)V_ * D_);

    // ---- embeddings ----
    const long long nX = (long long)B_ * S_ * D_;
    const long long nS = (long long)B_ * NB_ * D_;
    embed_kernel<<<blocks(nX), 256, 0, stream>>>(tok, inst, tokE, instE, x, nX);
    bar_kernel<<<blocks(nS), 256, 0, stream>>>(barE, sumx, nS);

    const int MR = B_ * S_;    // 2048
    const int MS = B_ * NB_;   // 128
    const int BH = B_ * H_;    // 16
    const float scale = 0.125f;   // DH^-0.5

    for (int l = 0; l < L_; ++l) {
        const _Float16* WqTl = WqT + (long long)l * DD;
        const _Float16* WkTl = WkT + (long long)l * DD;
        const _Float16* WvTl = WvT + (long long)l * DD;
        const _Float16* WoTl = WoT + (long long)l * DD;

        // attention LNs
        ln_kernel<<<MS, 128, 0, stream>>>(sumx, lnS, ln_as_g + l * D_, ln_as_b + l * D_);
        ln_kernel<<<MR, 128, 0, stream>>>(x,    lnR, ln_ar_g + l * D_, ln_ar_b + l * D_);

        // projections
        gemmF(lnS, WqTl, pSq, MS, D_, D_, D_, 0,0,0,0,0,0, 1, 1.f, 0, 1);
        gemmF(lnR, WqTl, pRq, MR, D_, D_, D_, 0,0,0,0,0,0, 1, 1.f, 0, 1);
        gemmF(lnS, WkTl, pSk, MS, D_, D_, D_, 0,0,0,0,0,0, 1, 1.f, 0, 1);
        gemmF(lnR, WkTl, pRk, MR, D_, D_, D_, 0,0,0,0,0,0, 1, 1.f, 0, 1);
        gemmF(lnS, WvTl, pSv, MS, D_, D_, D_, 0,0,0,0,0,0, 1, 1.f, 0, 1);
        gemmF(lnR, WvTl, pRv, MR, D_, D_, D_, 0,0,0,0,0,0, 1, 1.f, 0, 1);

        // RoPE + scatter into head-major f16 layouts
        rope_scatter_kernel<<<blocks(nX), 256, 0, stream>>>(pRq, qrH, S_, 0, S_, nX);
        rope_scatter_kernel<<<blocks(nS), 256, 0, stream>>>(pSq, qsH, NB_, 0, NB_, nS);
        rope_scatter_kernel<<<blocks(nS), 256, 0, stream>>>(pSk, kH, NB_, 0, T_, nS);
        rope_scatter_kernel<<<blocks(nX), 256, 0, stream>>>(pRk, kH, S_, NB_, T_, nX);
        vT_scatter_kernel<<<blocks(nS), 256, 0, stream>>>(pSv, vTH, NB_, 0, nS);
        vT_scatter_kernel<<<blocks(nX), 256, 0, stream>>>(pRv, vTH, S_, NB_, nX);

        // regular-stream attention
        gemmH(qrH, kH, scR, S_, T_, DH_, T_,
              (long long)S_ * DH_, 0, (long long)T_ * DH_, 0, (long long)S_ * T_, 0,
              1, scale, 0, BH);
        softmax_reg_kernel<<<dim3(S_, H_, B_), 128, 0, stream>>>(scR, pbR, chord);
        gemmH(pbR, vTH, aR, S_, DH_, T_, D_,
              (long long)H_ * S_ * T_, (long long)S_ * T_,
              (long long)H_ * DH_ * T_, (long long)DH_ * T_,
              (long long)S_ * D_, DH_, H_, 1.f, 0, BH);
        gemmF(aR, WoTl, x, MR, D_, D_, D_, 0,0,0,0,0,0, 1, 1.f, 1, 1);

        // summary-stream attention
        gemmH(qsH, kH, scS, NB_, T_, DH_, T_,
              (long long)NB_ * DH_, 0, (long long)T_ * DH_, 0, (long long)NB_ * T_, 0,
              1, scale, 0, BH);
        softmax_sum_kernel<<<dim3(NB_, H_, B_), 128, 0, stream>>>(scS, pbS, chord);
        gemmH(pbS, vTH, aS, NB_, DH_, T_, D_,
              (long long)H_ * NB_ * T_, (long long)NB_ * T_,
              (long long)H_ * DH_ * T_, (long long)DH_ * T_,
              (long long)NB_ * D_, DH_, H_, 1.f, 0, BH);
        gemmF(aS, WoTl, sumx, MS, D_, D_, D_, 0,0,0,0,0,0, 1, 1.f, 1, 1);

        // summary FFN (SwiGLU)
        ln_kernel<<<MS, 128, 0, stream>>>(sumx, lnS, ln_fs_g + l * D_, ln_fs_b + l * D_);
        gemmF(lnS, fsw1T + (long long)l * DF, f1S, MS, F_, D_, F_, 0,0,0,0,0,0, 1, 1.f, 0, 1);
        gemmF(lnS, fsw3T + (long long)l * DF, f3S, MS, F_, D_, F_, 0,0,0,0,0,0, 1, 1.f, 0, 1);
        silu_mul_kernel<<<blocks((long long)MS * F_), 256, 0, stream>>>(f1S, f3S, (long long)MS * F_);
        gemmF(f1S, fsw2T + (long long)l * DF, sumx, MS, D_, F_, D_, 0,0,0,0,0,0, 1, 1.f, 1, 1);

        // regular FFN (SwiGLU)
        ln_kernel<<<MR, 128, 0, stream>>>(x, lnR, ln_fr_g + l * D_, ln_fr_b + l * D_);
        gemmF(lnR, frw1T + (long long)l * DF, f1R, MR, F_, D_, F_, 0,0,0,0,0,0, 1, 1.f, 0, 1);
        gemmF(lnR, frw3T + (long long)l * DF, f3R, MR, F_, D_, F_, 0,0,0,0,0,0, 1, 1.f, 0, 1);
        silu_mul_kernel<<<blocks((long long)MR * F_), 256, 0, stream>>>(f1R, f3R, (long long)MR * F_);
        gemmF(f1R, frw2T + (long long)l * DF, x, MR, D_, F_, D_, 0,0,0,0,0,0, 1, 1.f, 1, 1);
    }

    // final LN + logits vs tok_emb^T
    ln_kernel<<<MR, 128, 0, stream>>>(x, lnR, out_g, out_b);
    gemmF(lnR, tokH, out, MR, V_, D_, V_, 0,0,0,0,0,0, 1, 1.f, 0, 1);
}
